// BertDiora_24721831756131
// MI455X (gfx1250) — compile-verified, matrix operand order checked by reading, not run on hardware
//
#include <hip/hip_runtime.h>
#include <hip/hip_bf16.h>
#include <math.h>
#include <stdint.h>

// ---------------------------------------------------------------------------
// BertDiora inside-outside chart on MI455X (gfx1250).
// Compute-bound (~5 TFLOP of 768-deep GEMMs, working set ~150MB < 192MB L2):
// charts kept in f16, GEMMs via v_wmma_f32_16x16x32_f16 (f32 accumulate),
// softmax / normalize / scores in f32. Weights pre-transposed to [n][k] f16.
// Global->LDS staging uses the CDNA5 async-to-LDS DMA path (ASYNCcnt) to
// avoid the VGPR round-trip; outputs drain via async-from-LDS stores.
// ---------------------------------------------------------------------------

typedef __attribute__((ext_vector_type(16))) _Float16 v16h;
typedef __attribute__((ext_vector_type(8)))  float    v8f;
typedef __attribute__((ext_vector_type(8)))  _Float16 h8;

#define N_LEN 32
#define BATCH 64
#define S     768
#define TRI   528          // (n+1)*n/2
#define ROWS  32           // batch rows per pair-kernel block
#define LDP   776          // padded LDS row stride (halves) for 768-wide rows
#define BSTR  40           // padded LDS stride for staged 32-K weight columns

__device__ __forceinline__ int tri_idx(int b, int e) {
    int Ln = e - b;
    return (N_LEN + 1) * N_LEN / 2
         - (N_LEN - Ln + 2) * (N_LEN - Ln + 1) / 2 + b;
}

// --- CDNA5 async Global<->LDS DMA (ISA 15.18.3 / 08_async_tensor) -----------
// LDS operand = low 32 bits of the generic pointer (HW uses addr[31:0] as the
// LDS byte address for the shared aperture, ISA 10.2).
__device__ __forceinline__ void async_load_b128(void* lds_dst, const void* gsrc) {
    unsigned l = (unsigned)(uintptr_t)lds_dst;
    unsigned long long g = (unsigned long long)(uintptr_t)gsrc;
    asm volatile("global_load_async_to_lds_b128 %0, %1, off"
                 :: "v"(l), "v"(g) : "memory");
}
__device__ __forceinline__ void async_store_b128(void* gdst, const void* lds_src) {
    unsigned l = (unsigned)(uintptr_t)lds_src;
    unsigned long long g = (unsigned long long)(uintptr_t)gdst;
    asm volatile("global_store_async_from_lds_b128 %1, %0, off"
                 :: "v"(l), "v"(g) : "memory");
}
__device__ __forceinline__ void wait_async() {
    asm volatile("s_wait_asynccnt 0x0" ::: "memory");
}

// A/B fragment for V_WMMA_F32_16X16X32_F16.
// Lane layout (ISA 7.12.2, 16-bit 16x32): major index = lane&15,
// K runs {kb..kb+7} and {kb+16..kb+23} with kb = (lane>>4)*8.
__device__ __forceinline__ v16h load_frag(const _Float16* p) {
    v16h f;
#pragma unroll
    for (int h = 0; h < 8; ++h) f[h]     = p[h];
#pragma unroll
    for (int h = 0; h < 8; ++h) f[h + 8] = p[16 + h];
    return f;
}

// Block-cooperative GEMM accumulate:
//   acc[mt] += A_lds(32 x kCount, stride LDP) @ W^T
// Wt laid out [n][K] f16 with row stride ldK halves; this wave owns the
// 16-column slice at (nWin + wave*16). kOff selects the K-offset inside a
// Wt row (used for the two W1 halves). Staging is async-to-LDS DMA,
// manually unrolled so both issues are straight-line (no exec masking).
__device__ __forceinline__ void gemm_acc(const _Float16* Albs,
                                         const _Float16* Wt, int ldK, int kOff,
                                         int kCount, int nWin,
                                         _Float16* Bstage,
                                         v8f acc[2], int lane, int tid) {
    const int kb = ((lane >> 4) << 3);
    const int wv = tid >> 5;
    const _Float16* a0 = Albs + (lane & 15) * LDP + kb;
    const _Float16* a1 = a0 + 16 * LDP;
    const _Float16* bb = Bstage + (wv * 16 + (lane & 15)) * BSTR + kb;
    // staging decomposition: thread handles items tid and tid+256 of 512
    const int c0  = tid >> 2,        k80 = (tid & 3) << 3;
    const int c1  = (tid + 256) >> 2, k81 = ((tid + 256) & 3) << 3;
    for (int kc = 0; kc < kCount; kc += 32) {
        __syncthreads();                       // previous Bstage consumed
        async_load_b128(Bstage + c0 * BSTR + k80,
                        Wt + (size_t)(nWin + c0) * ldK + kOff + kc + k80);
        async_load_b128(Bstage + c1 * BSTR + k81,
                        Wt + (size_t)(nWin + c1) * ldK + kOff + kc + k81);
        wait_async();                          // this wave's DMAs landed
        __syncthreads();                       // all waves' DMAs visible
        v16h fb  = load_frag(bb);
        v16h fa0 = load_frag(a0 + kc);
        acc[0] = __builtin_amdgcn_wmma_f32_16x16x32_f16(
            false, fa0, false, fb, (short)0, acc[0], false, false);
        v16h fa1 = load_frag(a1 + kc);
        acc[1] = __builtin_amdgcn_wmma_f32_16x16x32_f16(
            false, fa1, false, fb, (short)0, acc[1], false, false);
    }
}

// ---------------------------------------------------------------------------
// Utility kernels
// ---------------------------------------------------------------------------
__global__ __launch_bounds__(256) void cast_f2h_kernel(const float* __restrict__ src,
                                                       _Float16* __restrict__ dst,
                                                       size_t count) {
    size_t i = (size_t)blockIdx.x * blockDim.x + threadIdx.x;
    size_t stride = (size_t)gridDim.x * blockDim.x;
    for (; i < count; i += stride) dst[i] = (_Float16)src[i];
}

__global__ __launch_bounds__(256) void zero_f32_kernel(float* __restrict__ p, size_t count) {
    size_t i = (size_t)blockIdx.x * blockDim.x + threadIdx.x;
    size_t stride = (size_t)gridDim.x * blockDim.x;
    for (; i < count; i += stride) p[i] = 0.0f;
}

// Transpose + cast: dst[n*K + k] = (f16)src[k*768 + n].
// grid = (768/32, K/32), block = 256 (32x8).
__global__ __launch_bounds__(256) void transpose_cast_kernel(const float* __restrict__ src,
                                                             _Float16* __restrict__ dst,
                                                             int K) {
    __shared__ float tile[32][33];
    const int tx = threadIdx.x & 31;
    const int ty = threadIdx.x >> 5;          // 0..7
    const int n0 = blockIdx.x * 32;
    const int k0 = blockIdx.y * 32;
#pragma unroll
    for (int j = 0; j < 32; j += 8)
        tile[ty + j][tx] = src[(size_t)(k0 + ty + j) * S + n0 + tx];
    __syncthreads();
#pragma unroll
    for (int j = 0; j < 32; j += 8)
        dst[(size_t)(n0 + ty + j) * K + k0 + tx] = (_Float16)tile[tx][ty + j];
}

// ---------------------------------------------------------------------------
// Word projection: tmp = tanh(X @ W_word + b_word), 2048 rows (32 spans x 64 batch)
// ---------------------------------------------------------------------------
__global__ __launch_bounds__(256) void wordproj_kernel(const _Float16* __restrict__ Xh,
                                                       const _Float16* __restrict__ Wwt,
                                                       const float* __restrict__ bias,
                                                       float* __restrict__ tmpOut) {
    extern __shared__ _Float16 smem[];
    _Float16* Ab     = smem;                   // ROWS x LDP
    _Float16* Bstage = Ab + ROWS * LDP;        // 128 x BSTR
    const int tid  = threadIdx.x;
    const int lane = tid & 31;
    const int wv   = tid >> 5;
    const int row0 = blockIdx.x * ROWS;

#pragma unroll
    for (int rep = 0; rep < 12; ++rep) {       // 32 rows x 96 chunks / 256 thr
        int i  = tid + rep * 256;
        int rr = i / 96;
        int c8 = (i % 96) * 8;
        async_load_b128(Ab + rr * LDP + c8, Xh + (size_t)(row0 + rr) * S + c8);
    }
    wait_async();
    for (int win = 0; win < S; win += 128) {
        v8f acc[2] = {};
        gemm_acc(Ab, Wwt, S, 0, S, win, Bstage, acc, lane, tid);
        int n = win + wv * 16 + (lane & 15);
        float bv = bias[n];
        int mo = (lane >> 4) * 8;
#pragma unroll
        for (int mt = 0; mt < 2; ++mt)
#pragma unroll
            for (int v = 0; v < 8; ++v) {
                float x = tanhf(acc[mt][v] + bv);
                tmpOut[(size_t)(row0 + mt * 16 + mo + v) * S + n] = x;
            }
    }
}

__global__ __launch_bounds__(256) void wordnorm_kernel(const float* __restrict__ tmpIn,
                                                       _Float16* __restrict__ insV) {
    __shared__ float red[256];
    const int row = blockIdx.x;
    const int tid = threadIdx.x;
    float v[3];
    float ss = 0.0f;
#pragma unroll
    for (int j = 0; j < 3; ++j) {
        v[j] = tmpIn[(size_t)row * S + tid + 256 * j];
        ss += v[j] * v[j];
    }
    red[tid] = ss;
    __syncthreads();
    for (int st = 128; st > 0; st >>= 1) {
        if (tid < st) red[tid] += red[tid + st];
        __syncthreads();
    }
    float inv = 1.0f / fmaxf(sqrtf(red[0]), 1e-12f);
#pragma unroll
    for (int j = 0; j < 3; ++j)
        insV[(size_t)row * S + tid + 256 * j] = (_Float16)(v[j] * inv);
}

// ---------------------------------------------------------------------------
// Fused pair kernel: per (pair, 32-batch-rows) block computes
//   score_bil = l^T W_bil r  (WMMA + LDS f32 atomics)
//   vec       = relu(relu([l;r] W1 + b1) W2 + b2)   (H kept in LDS)
// mode 0: inside (l=left span, r=right span); mode 1: outside (l=parent, r=sister)
// ---------------------------------------------------------------------------
__global__ __launch_bounds__(256) void pair_kernel(int L, int mode,
        const _Float16* __restrict__ chartA, const _Float16* __restrict__ chartB,
        const float* __restrict__ scoreA,    const float* __restrict__ scoreB,
        const _Float16* __restrict__ Wbt,    const _Float16* __restrict__ W1t,
        const _Float16* __restrict__ W2t,
        const float* __restrict__ b1,        const float* __restrict__ b2,
        _Float16* __restrict__ pairVec,      float* __restrict__ pairScore) {
    extern __shared__ _Float16 smem[];
    _Float16* Lb     = smem;                      // ROWS x LDP
    _Float16* Rb     = Lb + ROWS * LDP;
    _Float16* Hb     = Rb + ROWS * LDP;
    _Float16* Bstage = Hb + ROWS * LDP;           // 128 x BSTR
    float* scoreAcc  = (float*)(Bstage + 128 * BSTR);   // 32 floats

    const int p    = blockIdx.x;
    const int r0   = blockIdx.y * ROWS;
    const int tid  = threadIdx.x;
    const int lane = tid & 31;
    const int wv   = tid >> 5;
    const int nbg  = N_LEN + 1 - L;

    int idxA, idxB;
    if (mode == 0) {
        int k = p / nbg + 1, b = p % nbg;
        idxA = tri_idx(b, b + k);
        idxB = tri_idx(b + k, b + L);
    } else {
        int q = p / nbg, b = p % nbg, e = b + L;
        if (q < b) { idxA = tri_idx(q, e); idxB = tri_idx(q, b); }
        else       { int c = q - b + e + 1; idxA = tri_idx(b, c); idxB = tri_idx(e, c); }
    }

    if (tid < ROWS) scoreAcc[tid] = 0.0f;
    // async DMA the two 32x768 chart tiles into LDS (straight-line issue)
#pragma unroll
    for (int rep = 0; rep < 12; ++rep) {
        int i  = tid + rep * 256;
        int rr = i / 96;
        int c8 = (i % 96) * 8;
        async_load_b128(Lb + rr * LDP + c8,
                        chartA + ((size_t)idxA * BATCH + r0 + rr) * S + c8);
        async_load_b128(Rb + rr * LDP + c8,
                        chartB + ((size_t)idxB * BATCH + r0 + rr) * S + c8);
    }
    wait_async();
    // (gemm_acc's leading __syncthreads makes tiles visible block-wide)

    // ---- bilinear: u = L @ W_bil, score += sum_n u[m,n] * R[m,n] ----
    for (int win = 0; win < S; win += 128) {
        v8f acc[2] = {};
        gemm_acc(Lb, Wbt, S, 0, S, win, Bstage, acc, lane, tid);
        int n  = win + wv * 16 + (lane & 15);
        int mo = (lane >> 4) * 8;
#pragma unroll
        for (int mt = 0; mt < 2; ++mt)
#pragma unroll
            for (int v = 0; v < 8; ++v) {
                int m = mt * 16 + mo + v;
                atomicAdd(&scoreAcc[m], acc[mt][v] * (float)Rb[m * LDP + n]);
            }
    }

    // ---- H = relu(L @ W1[0:768] + R @ W1[768:1536] + b1), kept in LDS ----
    for (int win = 0; win < S; win += 128) {
        v8f acc[2] = {};
        gemm_acc(Lb, W1t, 2 * S, 0, S, win, Bstage, acc, lane, tid);
        gemm_acc(Rb, W1t, 2 * S, S, S, win, Bstage, acc, lane, tid);
        int n  = win + wv * 16 + (lane & 15);
        float bv = b1[n];
        int mo = (lane >> 4) * 8;
#pragma unroll
        for (int mt = 0; mt < 2; ++mt)
#pragma unroll
            for (int v = 0; v < 8; ++v) {
                float x = acc[mt][v] + bv;
                Hb[(mt * 16 + mo + v) * LDP + n] = (_Float16)(x > 0.0f ? x : 0.0f);
            }
    }
    __syncthreads();   // H complete; Lb is dead from here on -> reuse as V stage

    // ---- V = relu(H @ W2 + b2) -> stage into Lb, then async store ----
    for (int win = 0; win < S; win += 128) {
        v8f acc[2] = {};
        gemm_acc(Hb, W2t, S, 0, S, win, Bstage, acc, lane, tid);
        int n  = win + wv * 16 + (lane & 15);
        float bv = b2[n];
        int mo = (lane >> 4) * 8;
#pragma unroll
        for (int mt = 0; mt < 2; ++mt)
#pragma unroll
            for (int v = 0; v < 8; ++v) {
                float x = acc[mt][v] + bv;
                Lb[(mt * 16 + mo + v) * LDP + n] = (_Float16)(x > 0.0f ? x : 0.0f);
            }
    }
    __syncthreads();   // V tile complete in LDS

    // drain V tile to pairVec with async LDS->global 16B stores
#pragma unroll
    for (int rep = 0; rep < 12; ++rep) {
        int i  = tid + rep * 256;
        int rr = i / 96;
        int c8 = (i % 96) * 8;
        async_store_b128(pairVec + ((size_t)p * BATCH + r0 + rr) * S + c8,
                         Lb + rr * LDP + c8);
    }

    if (tid < ROWS) {
        int m = tid;
        float s = scoreAcc[m]
                + scoreA[(size_t)idxA * BATCH + r0 + m]
                + scoreB[(size_t)idxB * BATCH + r0 + m];
        pairScore[(size_t)p * BATCH + r0 + m] = s;
    }
    wait_async();      // async stores complete before wave exit
}

// ---------------------------------------------------------------------------
// Combine: softmax over split axis, weighted sums, normalize, write charts.
// grid = (numBegins, BATCH)
// ---------------------------------------------------------------------------
__global__ __launch_bounds__(256) void combine_kernel(int L, int mode,
        const _Float16* __restrict__ pairVec, const float* __restrict__ pairScore,
        _Float16* __restrict__ chartV,        float* __restrict__ chartS) {
    __shared__ float red[256];
    const int b   = blockIdx.x;
    const int r   = blockIdx.y;
    const int tid = threadIdx.x;
    const int nbg = N_LEN + 1 - L;
    const int nk  = (mode == 0) ? (L - 1) : (N_LEN - L);
    const int tgt = tri_idx(b, b + L);

    float mx = -3.0e38f;
    for (int k = 0; k < nk; ++k)
        mx = fmaxf(mx, pairScore[(size_t)(k * nbg + b) * BATCH + r]);
    float den = 0.0f, num = 0.0f;
    for (int k = 0; k < nk; ++k) {
        float s = pairScore[(size_t)(k * nbg + b) * BATCH + r];
        float e = __expf(s - mx);
        den += e;
        num += e * s;
    }
    float inv_den  = 1.0f / den;
    float outScore = num * inv_den;

    float acc[3] = {0.0f, 0.0f, 0.0f};
    for (int k = 0; k < nk; ++k) {
        float w = __expf(pairScore[(size_t)(k * nbg + b) * BATCH + r] - mx) * inv_den;
        const _Float16* pv = pairVec + ((size_t)(k * nbg + b) * BATCH + r) * S;
#pragma unroll
        for (int j = 0; j < 3; ++j)
            acc[j] += w * (float)pv[tid + 256 * j];
    }
    float ss = acc[0] * acc[0] + acc[1] * acc[1] + acc[2] * acc[2];
    red[tid] = ss;
    __syncthreads();
    for (int st = 128; st > 0; st >>= 1) {
        if (tid < st) red[tid] += red[tid + st];
        __syncthreads();
    }
    float inv = 1.0f / fmaxf(sqrtf(red[0]), 1e-12f);
    _Float16* dst = chartV + ((size_t)tgt * BATCH + r) * S;
#pragma unroll
    for (int j = 0; j < 3; ++j)
        dst[tid + 256 * j] = (_Float16)(acc[j] * inv);
    if (tid == 0) chartS[(size_t)tgt * BATCH + r] = outScore;
}

// ---------------------------------------------------------------------------
// Outside root init: outV[tri(0,n)] = broadcast(normalize(root_bias))
// ---------------------------------------------------------------------------
__global__ __launch_bounds__(256) void root_kernel(const float* __restrict__ root_bias,
                                                   _Float16* __restrict__ outV) {
    __shared__ float red[256];
    const int tid = threadIdx.x;
    float v[3];
    float ss = 0.0f;
#pragma unroll
    for (int j = 0; j < 3; ++j) {
        v[j] = root_bias[tid + 256 * j];
        ss += v[j] * v[j];
    }
    red[tid] = ss;
    __syncthreads();
    for (int st = 128; st > 0; st >>= 1) {
        if (tid < st) red[tid] += red[tid + st];
        __syncthreads();
    }
    float inv = 1.0f / fmaxf(sqrtf(red[0]), 1e-12f);
    const size_t base = (size_t)tri_idx(0, N_LEN) * BATCH * S;
    for (int r = 0; r < BATCH; ++r)
#pragma unroll
        for (int j = 0; j < 3; ++j)
            outV[base + (size_t)r * S + tid + 256 * j] = (_Float16)(v[j] * inv);
}

// ---------------------------------------------------------------------------
// Final: out[i*64+r] = 1 - cos(terminal, base_vecs)
// ---------------------------------------------------------------------------
__global__ __launch_bounds__(256) void cosine_kernel(const _Float16* __restrict__ outV,
                                                     const float* __restrict__ base,
                                                     float* __restrict__ out) {
    __shared__ float r1[256], r2[256], r3[256];
    const int bid = blockIdx.x;   // 0..2047 = span*64 + batch (rows 0..31 of chart)
    const int tid = threadIdx.x;
    const _Float16* t = outV + (size_t)bid * S;
    const float*    x = base + (size_t)bid * S;
    float dot = 0.0f, nt = 0.0f, nb = 0.0f;
#pragma unroll
    for (int j = 0; j < 3; ++j) {
        int c = tid + 256 * j;
        float tv = (float)t[c];
        float xv = x[c];
        dot += tv * xv;
        nt  += tv * tv;
        nb  += xv * xv;
    }
    r1[tid] = dot; r2[tid] = nt; r3[tid] = nb;
    __syncthreads();
    for (int st = 128; st > 0; st >>= 1) {
        if (tid < st) { r1[tid] += r1[tid + st]; r2[tid] += r2[tid + st]; r3[tid] += r3[tid + st]; }
        __syncthreads();
    }
    if (tid == 0)
        out[bid] = 1.0f - r1[0] / (fmaxf(sqrtf(r2[0]), 1e-8f) * fmaxf(sqrtf(r3[0]), 1e-8f));
}

// ---------------------------------------------------------------------------
// Host-side orchestration
// ---------------------------------------------------------------------------
extern "C" void kernel_launch(void* const* d_in, const int* in_sizes, int n_in,
                              void* d_out, int out_size, void* d_ws, size_t ws_size,
                              hipStream_t stream) {
    (void)in_sizes; (void)n_in; (void)out_size; (void)ws_size;
    const float* base_vecs = (const float*)d_in[0];   // (32, 64, 768)
    const float* W_word    = (const float*)d_in[1];   // (768, 768)
    const float* b_word    = (const float*)d_in[2];   // (768,)
    const float* W_bil     = (const float*)d_in[3];   // (768, 768)
    const float* W1        = (const float*)d_in[4];   // (1536, 768)
    const float* b1        = (const float*)d_in[5];   // (768,)
    const float* W2        = (const float*)d_in[6];   // (768, 768)
    const float* b2        = (const float*)d_in[7];   // (768,)
    const float* root_bias = (const float*)d_in[8];   // (768,)
    float* out = (float*)d_out;                       // (32, 64)

    // workspace carve-out (~216 MB total)
    char* ws = (char*)d_ws;
    auto carve = [&](size_t bytes) -> char* {
        char* p = ws;
        ws += (bytes + 255) & ~(size_t)255;
        return p;
    };
    _Float16* insV   = (_Float16*)carve((size_t)TRI * BATCH * S * 2);
    _Float16* outV   = (_Float16*)carve((size_t)TRI * BATCH * S * 2);
    float*    insS   = (float*)   carve((size_t)TRI * BATCH * 4);
    float*    outS   = (float*)   carve((size_t)TRI * BATCH * 4);
    _Float16* Wwt    = (_Float16*)carve((size_t)S * S * 2);        // W_word^T  [n][k]
    _Float16* Wbt    = (_Float16*)carve((size_t)S * S * 2);        // W_bil^T   [n][k]
    _Float16* W1t    = (_Float16*)carve((size_t)2 * S * S * 2);    // W1^T      [n][2k]
    _Float16* W2t    = (_Float16*)carve((size_t)S * S * 2);        // W2^T      [n][k]
    _Float16* bv_h   = (_Float16*)carve((size_t)N_LEN * BATCH * S * 2);
    float*    tmpF   = (float*)   carve((size_t)N_LEN * BATCH * S * 4);
    const int MAXP   = (N_LEN) * (N_LEN - 1);   // 992 max pairs (outside L=1)
    _Float16* pairV  = (_Float16*)carve((size_t)MAXP * BATCH * S * 2);
    float*    pairS  = (float*)   carve((size_t)MAXP * BATCH * 4);

    const size_t GEMM_SMEM = ((size_t)ROWS * LDP + 128 * BSTR) * 2;               // ~59 KB
    const size_t PAIR_SMEM = ((size_t)3 * ROWS * LDP + 128 * BSTR) * 2 + ROWS * 4; // ~156 KB

    // weight transpose+cast (once per launch), activation cast, zero-init
    transpose_cast_kernel<<<dim3(S / 32, S / 32), 256, 0, stream>>>(W_word, Wwt, S);
    transpose_cast_kernel<<<dim3(S / 32, S / 32), 256, 0, stream>>>(W_bil,  Wbt, S);
    transpose_cast_kernel<<<dim3(S / 32, 2 * S / 32), 256, 0, stream>>>(W1, W1t, 2 * S);
    transpose_cast_kernel<<<dim3(S / 32, S / 32), 256, 0, stream>>>(W2,     W2t, S);
    cast_f2h_kernel<<<512, 256, 0, stream>>>(base_vecs, bv_h, (size_t)N_LEN * BATCH * S);
    zero_f32_kernel<<<64, 256, 0, stream>>>(insS, (size_t)TRI * BATCH);
    zero_f32_kernel<<<64, 256, 0, stream>>>(outS, (size_t)TRI * BATCH);

    // word projection -> normalized length-1 inside vectors
    wordproj_kernel<<<(N_LEN * BATCH) / ROWS, 256, GEMM_SMEM, stream>>>(bv_h, Wwt, b_word, tmpF);
    wordnorm_kernel<<<N_LEN * BATCH, 256, 0, stream>>>(tmpF, insV);

    // inside pass
    for (int L = 2; L <= N_LEN; ++L) {
        int nbg = N_LEN + 1 - L;
        int np  = (L - 1) * nbg;
        pair_kernel<<<dim3(np, BATCH / ROWS), 256, PAIR_SMEM, stream>>>(
            L, 0, insV, insV, insS, insS, Wbt, W1t, W2t, b1, b2, pairV, pairS);
        combine_kernel<<<dim3(nbg, BATCH), 256, 0, stream>>>(
            L, 0, pairV, pairS, insV, insS);
    }

    // outside pass
    root_kernel<<<1, 256, 0, stream>>>(root_bias, outV);
    for (int L = N_LEN - 1; L >= 1; --L) {
        int nbg = N_LEN + 1 - L;
        int np  = (N_LEN - L) * nbg;
        pair_kernel<<<dim3(np, BATCH / ROWS), 256, PAIR_SMEM, stream>>>(
            L, 1, outV, insV, outS, insS, Wbt, W1t, W2t, b1, b2, pairV, pairS);
        combine_kernel<<<dim3(nbg, BATCH), 256, 0, stream>>>(
            L, 1, pairV, pairS, outV, outS);
    }

    // final cosine distance on terminal outside vectors
    cosine_kernel<<<N_LEN * BATCH, 256, 0, stream>>>(outV, base_vecs, out);
}
